// ModelInternalClassSelection_55611236549192
// MI455X (gfx1250) — compile-verified, matrix operand order checked by reading, not run on hardware
//
#include <hip/hip_runtime.h>
#include <hip/hip_bf16.h>

typedef __bf16 bf16;
typedef bf16 v16bf __attribute__((ext_vector_type(16)));
typedef bf16 bf16x8 __attribute__((ext_vector_type(8)));
typedef bf16 bf16x4 __attribute__((ext_vector_type(4)));
typedef float v8f __attribute__((ext_vector_type(8)));

#define BDIM 8192
#define CIN  2048
#define HDIM 1024
#define GDIM 512
#define EDIM 16
#define DDIM 512

__device__ __forceinline__ v8f wmma_bf16(v16bf a, v16bf b, v8f c) {
  // D = A*B + C, 16x16x32 bf16 -> f32 accum
  return __builtin_amdgcn_wmma_f32_16x16x32_bf16(false, a, false, b, (short)0, c,
                                                 false, false);
}

__device__ __forceinline__ v16bf lds_frag(const bf16* p) {
  union { v16bf v; bf16x8 h[2]; } u;
  u.h[0] = *(const bf16x8*)p;
  u.h[1] = *(const bf16x8*)(p + 8);
  return u.v;
}

__device__ __forceinline__ void zero_acc(v8f acc[4][2]) {
  for (int i = 0; i < 4; i++)
    for (int j = 0; j < 2; j++)
      for (int q = 0; q < 8; q++) acc[i][j][q] = 0.0f;
}

// ---------------------------------------------------------------------------
// One-time weight prep: dst[batch][c][r] = bf16(src[batch][r][c])
// 32x32 tiles through LDS; coalesced f32 reads, coalesced bf16 writes.
// ---------------------------------------------------------------------------
__global__ __launch_bounds__(256) void transpose_cvt(
    const float* __restrict__ src, bf16* __restrict__ dst, int R, int C) {
  __shared__ float T[32][33];
  const int t = threadIdx.x;
  const int tx = t & 31;
  const int ty = t >> 5;  // 0..7
  const size_t base = (size_t)blockIdx.z * R * C;
  const int r0 = blockIdx.y * 32;
  const int c0 = blockIdx.x * 32;
  for (int j = 0; j < 4; j++)
    T[ty + j * 8][tx] = src[base + (size_t)(r0 + ty + j * 8) * C + c0 + tx];
  __syncthreads();
  for (int j = 0; j < 4; j++)
    dst[base + (size_t)(c0 + ty + j * 8) * R + r0 + tx] =
        (bf16)T[tx][ty + j * 8];
}

// ---------------------------------------------------------------------------
// 128x128 GEMM tile, K-loop of 64, software-pipelined (register prefetch).
// A: M x K (f32 or bf16, row-major).  Bt: N x K bf16 (pre-transposed).
// C = act(A @ B + bias) -> bf16.
// ---------------------------------------------------------------------------
template <bool A_F32, bool RELU>
__global__ __launch_bounds__(256) void gemm_wmma(const void* __restrict__ Ap,
                                                 const bf16* __restrict__ Bt,
                                                 const float* __restrict__ bias,
                                                 bf16* __restrict__ Cout,
                                                 int M, int K, int N) {
  __shared__ bf16 As[128][72];  // [m][k], pad -> conflict-free frag loads
  __shared__ bf16 Bs[128][72];  // [n][k]

  const int t = threadIdx.x;
  const int lane = t & 31;
  const int wid = t >> 5;
  const int wm = (wid >> 2) * 64;  // wave row offset (0/64)
  const int wn = (wid & 3) * 32;   // wave col offset (0..96)
  const int lrow = lane & 15;
  const int khalf = (lane >> 4) * 16;
  const int m0 = blockIdx.y * 128;
  const int n0 = blockIdx.x * 128;
  const int c8 = (t & 7) * 8;   // bf16 staging: 8 chunks/row, 32 rows/pass
  const int r0 = t >> 3;
  const int c4 = (t & 15) * 4;  // f32 staging: 16 quads/row, 16 rows/pass
  const int rA = t >> 4;

  bf16x8 aR[4], bR[4];
  float4 aF[8];

  // ---- prologue: fetch + stage tile 0 ----
  if (A_F32) {
    const float* A = (const float*)Ap;
    for (int i = 0; i < 8; i++)
      aF[i] = *(const float4*)(A + (size_t)(m0 + rA + i * 16) * K + c4);
  } else {
    const bf16* A = (const bf16*)Ap;
    for (int i = 0; i < 4; i++)
      aR[i] = *(const bf16x8*)(A + (size_t)(m0 + r0 + i * 32) * K + c8);
  }
  for (int i = 0; i < 4; i++)
    bR[i] = *(const bf16x8*)(Bt + (size_t)(n0 + r0 + i * 32) * K + c8);

  if (A_F32) {
    for (int i = 0; i < 8; i++) {
      bf16x4 o;
      o[0] = (bf16)aF[i].x; o[1] = (bf16)aF[i].y;
      o[2] = (bf16)aF[i].z; o[3] = (bf16)aF[i].w;
      *(bf16x4*)&As[rA + i * 16][c4] = o;
    }
  } else {
    for (int i = 0; i < 4; i++) *(bf16x8*)&As[r0 + i * 32][c8] = aR[i];
  }
  for (int i = 0; i < 4; i++) *(bf16x8*)&Bs[r0 + i * 32][c8] = bR[i];
  __syncthreads();

  v8f acc[4][2];
  zero_acc(acc);

  for (int kt = 0; kt < K; kt += 64) {
    const int kn = kt + 64;
    // ---- prefetch next tile into registers (latency overlapped w/ WMMA) ----
    if (kn < K) {
      if (A_F32) {
        const float* A = (const float*)Ap;
        for (int i = 0; i < 8; i++)
          aF[i] = *(const float4*)(A + (size_t)(m0 + rA + i * 16) * K + kn + c4);
      } else {
        const bf16* A = (const bf16*)Ap;
        for (int i = 0; i < 4; i++)
          aR[i] = *(const bf16x8*)(A + (size_t)(m0 + r0 + i * 32) * K + kn + c8);
      }
      for (int i = 0; i < 4; i++)
        bR[i] = *(const bf16x8*)(Bt + (size_t)(n0 + r0 + i * 32) * K + kn + c8);
    }
    // ---- compute on current LDS tile ----
    for (int ks = 0; ks < 64; ks += 32) {
      v16bf a[4], b[2];
      for (int tm = 0; tm < 4; tm++)
        a[tm] = lds_frag(&As[wm + tm * 16 + lrow][ks + khalf]);
      for (int tn = 0; tn < 2; tn++)
        b[tn] = lds_frag(&Bs[wn + tn * 16 + lrow][ks + khalf]);
      for (int tm = 0; tm < 4; tm++)
        for (int tn = 0; tn < 2; tn++)
          acc[tm][tn] = wmma_bf16(a[tm], b[tn], acc[tm][tn]);
    }
    __syncthreads();
    // ---- commit prefetched tile to LDS ----
    if (kn < K) {
      if (A_F32) {
        for (int i = 0; i < 8; i++) {
          bf16x4 o;
          o[0] = (bf16)aF[i].x; o[1] = (bf16)aF[i].y;
          o[2] = (bf16)aF[i].z; o[3] = (bf16)aF[i].w;
          *(bf16x4*)&As[rA + i * 16][c4] = o;
        }
      } else {
        for (int i = 0; i < 4; i++) *(bf16x8*)&As[r0 + i * 32][c8] = aR[i];
      }
      for (int i = 0; i < 4; i++) *(bf16x8*)&Bs[r0 + i * 32][c8] = bR[i];
      __syncthreads();
    }
  }

  // ---- epilogue: bias + relu -> bf16 global ----
  for (int tm = 0; tm < 4; tm++) {
    const int rbase = m0 + wm + tm * 16 + (lane >> 4) * 8;
    for (int tn = 0; tn < 2; tn++) {
      const int col = n0 + wn + tn * 16 + (lane & 15);
      const float bv = bias[col];
      for (int r = 0; r < 8; r++) {
        float v = acc[tm][tn][r] + bv;
        if (RELU) v = fmaxf(v, 0.0f);
        Cout[(size_t)(rbase + r) * N + col] = (bf16)v;
      }
    }
  }
}

// ---------------------------------------------------------------------------
// Fused expert: per (batch-tile, expert) WG.
//   H1 = relu(x1_tile @ We1[e] + be1[e])  -> 128x512 bf16 kept in LDS
//   V  = H1 @ We2[e] + be2[e]             -> strided store to vs[b, d, e]
// ~170 KB of the 320 KB WGP LDS; the 268 MB 'h' tensor never touches HBM.
// Weights pre-transposed bf16: We1t [E][D][H], We2t [E][D][D].
// ---------------------------------------------------------------------------
__global__ __launch_bounds__(256) void expert_fused(
    const bf16* __restrict__ x1, const bf16* __restrict__ We1t,
    const float* __restrict__ be1, const bf16* __restrict__ We2t,
    const float* __restrict__ be2, float* __restrict__ vs) {
  __shared__ bf16 Hs[128][520];  // full 128x512 H1 tile (+pad)
  __shared__ bf16 As[128][72];
  __shared__ bf16 Bs[128][72];

  const int t = threadIdx.x;
  const int lane = t & 31;
  const int wid = t >> 5;
  const int wm = (wid >> 2) * 64;
  const int wn = (wid & 3) * 32;
  const int lrow = lane & 15;
  const int khalf = (lane >> 4) * 16;
  const int e = blockIdx.x;
  const int m0 = blockIdx.y * 128;
  const int c8 = (t & 7) * 8;
  const int r0 = t >> 3;

  const bf16* W1e = We1t + (size_t)e * DDIM * HDIM;  // [D][H]
  const bf16* W2e = We2t + (size_t)e * DDIM * DDIM;  // [D][D]
  const float* b1e = be1 + e * DDIM;
  const float* b2e = be2 + e * DDIM;

  bf16x8 aR[4], bR[4];

  // ---------------- phase 1: H1 = relu(x1 @ We1[e] + be1[e]) ----------------
  for (int nt = 0; nt < 4; nt++) {
    const int n0 = nt * 128;
    // prologue tile 0
    for (int i = 0; i < 4; i++)
      aR[i] = *(const bf16x8*)(x1 + (size_t)(m0 + r0 + i * 32) * HDIM + c8);
    for (int i = 0; i < 4; i++)
      bR[i] = *(const bf16x8*)(W1e + (size_t)(n0 + r0 + i * 32) * HDIM + c8);
    for (int i = 0; i < 4; i++) *(bf16x8*)&As[r0 + i * 32][c8] = aR[i];
    for (int i = 0; i < 4; i++) *(bf16x8*)&Bs[r0 + i * 32][c8] = bR[i];
    __syncthreads();

    v8f acc[4][2];
    zero_acc(acc);
    for (int kt = 0; kt < HDIM; kt += 64) {
      const int kn = kt + 64;
      if (kn < HDIM) {
        for (int i = 0; i < 4; i++)
          aR[i] = *(const bf16x8*)(x1 + (size_t)(m0 + r0 + i * 32) * HDIM + kn + c8);
        for (int i = 0; i < 4; i++)
          bR[i] = *(const bf16x8*)(W1e + (size_t)(n0 + r0 + i * 32) * HDIM + kn + c8);
      }
      for (int ks = 0; ks < 64; ks += 32) {
        v16bf a[4], b[2];
        for (int tm = 0; tm < 4; tm++)
          a[tm] = lds_frag(&As[wm + tm * 16 + lrow][ks + khalf]);
        for (int tn = 0; tn < 2; tn++)
          b[tn] = lds_frag(&Bs[wn + tn * 16 + lrow][ks + khalf]);
        for (int tm = 0; tm < 4; tm++)
          for (int tn = 0; tn < 2; tn++)
            acc[tm][tn] = wmma_bf16(a[tm], b[tn], acc[tm][tn]);
      }
      __syncthreads();
      if (kn < HDIM) {
        for (int i = 0; i < 4; i++) *(bf16x8*)&As[r0 + i * 32][c8] = aR[i];
        for (int i = 0; i < 4; i++) *(bf16x8*)&Bs[r0 + i * 32][c8] = bR[i];
        __syncthreads();
      }
    }
    // epilogue into LDS H1 tile
    for (int tm = 0; tm < 4; tm++) {
      const int rbase = wm + tm * 16 + (lane >> 4) * 8;
      for (int tn = 0; tn < 2; tn++) {
        const int col = n0 + wn + tn * 16 + (lane & 15);
        const float bv = b1e[col];
        for (int r = 0; r < 8; r++) {
          float v = acc[tm][tn][r] + bv;
          v = fmaxf(v, 0.0f);
          Hs[rbase + r][col] = (bf16)v;
        }
      }
    }
    __syncthreads();
  }

  // ---------------- phase 2: V = H1 @ We2[e] + be2[e] ----------------
  for (int nt = 0; nt < 4; nt++) {
    const int n0 = nt * 128;
    for (int i = 0; i < 4; i++)
      bR[i] = *(const bf16x8*)(W2e + (size_t)(n0 + r0 + i * 32) * DDIM + c8);
    for (int i = 0; i < 4; i++) *(bf16x8*)&Bs[r0 + i * 32][c8] = bR[i];
    __syncthreads();

    v8f acc[4][2];
    zero_acc(acc);
    for (int kt = 0; kt < DDIM; kt += 64) {
      const int kn = kt + 64;
      if (kn < DDIM) {
        for (int i = 0; i < 4; i++)
          bR[i] = *(const bf16x8*)(W2e + (size_t)(n0 + r0 + i * 32) * DDIM + kn + c8);
      }
      for (int ks = 0; ks < 64; ks += 32) {
        v16bf a[4], b[2];
        for (int tm = 0; tm < 4; tm++)
          a[tm] = lds_frag(&Hs[wm + tm * 16 + lrow][kt + ks + khalf]);
        for (int tn = 0; tn < 2; tn++)
          b[tn] = lds_frag(&Bs[wn + tn * 16 + lrow][ks + khalf]);
        for (int tm = 0; tm < 4; tm++)
          for (int tn = 0; tn < 2; tn++)
            acc[tm][tn] = wmma_bf16(a[tm], b[tn], acc[tm][tn]);
      }
      __syncthreads();
      if (kn < DDIM) {
        for (int i = 0; i < 4; i++) *(bf16x8*)&Bs[r0 + i * 32][c8] = bR[i];
        __syncthreads();
      }
    }
    // epilogue: vs_stacked layout [B, D, E]
    for (int tm = 0; tm < 4; tm++) {
      const int rbase = m0 + wm + tm * 16 + (lane >> 4) * 8;
      for (int tn = 0; tn < 2; tn++) {
        const int col = n0 + wn + tn * 16 + (lane & 15);
        const float bv = b2e[col];
        for (int r = 0; r < 8; r++) {
          const float v = acc[tm][tn][r] + bv;
          vs[((size_t)(rbase + r) * DDIM + col) * EDIM + e] = v;
        }
      }
    }
  }
}

// ---------------------------------------------------------------------------
// Gating: logits = g @ Wg2 + bg2, softmax over E=16. One wave per row.
// ---------------------------------------------------------------------------
__global__ __launch_bounds__(256) void gating_softmax(
    const bf16* __restrict__ g, const float* __restrict__ Wg2,
    const float* __restrict__ bg2, float* __restrict__ wsout) {
  __shared__ float Wt[EDIM][GDIM];  // transposed: [e][k], 32 KB
  const int t = threadIdx.x;
  for (int idx = t; idx < GDIM * EDIM; idx += 256) {
    const int k = idx >> 4;
    const int e2 = idx & 15;
    Wt[e2][k] = Wg2[idx];
  }
  __syncthreads();

  const int lane = t & 31;
  const int wid = t >> 5;
  const int row = blockIdx.x * 8 + wid;

  float s[EDIM];
  for (int e2 = 0; e2 < EDIM; e2++) s[e2] = 0.0f;
  for (int k = lane; k < GDIM; k += 32) {
    const float gv = (float)g[(size_t)row * GDIM + k];
    for (int e2 = 0; e2 < EDIM; e2++) s[e2] += gv * Wt[e2][k];
  }
  for (int off = 16; off > 0; off >>= 1)
    for (int e2 = 0; e2 < EDIM; e2++) s[e2] += __shfl_xor(s[e2], off, 32);

  float mx = -1e30f;
  for (int e2 = 0; e2 < EDIM; e2++) {
    s[e2] += bg2[e2];
    mx = fmaxf(mx, s[e2]);
  }
  float den = 0.0f;
  for (int e2 = 0; e2 < EDIM; e2++) {
    s[e2] = __expf(s[e2] - mx);
    den += s[e2];
  }
  const float inv = 1.0f / den;
  if (lane == 0)
    for (int e2 = 0; e2 < EDIM; e2++)
      wsout[(size_t)row * EDIM + e2] = s[e2] * inv;
}

// ---------------------------------------------------------------------------
// out[b,d] = sum_e ws[b,e] * vs[b,d,e]   (deterministic, no atomics)
// ---------------------------------------------------------------------------
__global__ __launch_bounds__(256) void gate_reduce(const float* __restrict__ wsg,
                                                   const float* __restrict__ vs,
                                                   float* __restrict__ out) {
  const int idx = blockIdx.x * 256 + threadIdx.x;  // b*D + d
  const int b = idx >> 9;                          // / DDIM
  const float4* v4 = (const float4*)(vs + (size_t)idx * EDIM);
  const float4* w4 = (const float4*)(wsg + (size_t)b * EDIM);
  float acc = 0.0f;
  for (int j = 0; j < 4; j++) {
    const float4 w = w4[j];
    const float4 v = v4[j];
    acc += w.x * v.x + w.y * v.y + w.z * v.z + w.w * v.w;
  }
  out[idx] = acc;
}

extern "C" void kernel_launch(void* const* d_in, const int* in_sizes, int n_in,
                              void* d_out, int out_size, void* d_ws,
                              size_t ws_size, hipStream_t stream) {
  (void)in_sizes; (void)n_in; (void)out_size; (void)ws_size;
  const float* x   = (const float*)d_in[0];
  const float* W1  = (const float*)d_in[1];
  const float* b1  = (const float*)d_in[2];
  const float* Wg1 = (const float*)d_in[3];
  const float* bg1 = (const float*)d_in[4];
  const float* Wg2 = (const float*)d_in[5];
  const float* bg2 = (const float*)d_in[6];
  const float* We1 = (const float*)d_in[7];
  const float* be1 = (const float*)d_in[8];
  const float* We2 = (const float*)d_in[9];
  const float* be2 = (const float*)d_in[10];

  float* out = (float*)d_out;                       // [B, D]
  float* wsg = out + (size_t)BDIM * DDIM;           // [B, E]
  float* vs  = wsg + (size_t)BDIM * EDIM;           // [B, D, E]

  bf16* x1b  = (bf16*)d_ws;                          // [B, H]
  bf16* gbuf = x1b  + (size_t)BDIM * HDIM;           // [B, G]
  bf16* W1t  = gbuf + (size_t)BDIM * GDIM;           // [H, CIN]
  bf16* Wg1t = W1t  + (size_t)HDIM * CIN;            // [G, H]
  bf16* We1t = Wg1t + (size_t)GDIM * HDIM;           // [E, D, H]
  bf16* We2t = We1t + (size_t)EDIM * DDIM * HDIM;    // [E, D, D]

  dim3 blk(256);
  // one-time weight transpose + f32->bf16 conversion
  transpose_cvt<<<dim3(HDIM / 32, CIN / 32, 1), blk, 0, stream>>>(W1, W1t, CIN, HDIM);
  transpose_cvt<<<dim3(GDIM / 32, HDIM / 32, 1), blk, 0, stream>>>(Wg1, Wg1t, HDIM, GDIM);
  transpose_cvt<<<dim3(DDIM / 32, HDIM / 32, EDIM), blk, 0, stream>>>(We1, We1t, HDIM, DDIM);
  transpose_cvt<<<dim3(DDIM / 32, DDIM / 32, EDIM), blk, 0, stream>>>(We2, We2t, DDIM, DDIM);

  // x1 = relu(x @ W1 + b1)
  gemm_wmma<true, true><<<dim3(HDIM / 128, BDIM / 128), blk, 0, stream>>>(
      x, W1t, b1, x1b, BDIM, CIN, HDIM);
  // g = relu(x1 @ Wg1 + bg1)
  gemm_wmma<false, true><<<dim3(GDIM / 128, BDIM / 128), blk, 0, stream>>>(
      x1b, Wg1t, bg1, gbuf, BDIM, HDIM, GDIM);
  // ws = softmax(g @ Wg2 + bg2)
  gating_softmax<<<dim3(BDIM / 8), blk, 0, stream>>>(gbuf, Wg2, bg2, wsg);
  // vs[b,d,e] = (relu(x1 @ We1[e] + be1) @ We2[e]) + be2   (fused in LDS)
  expert_fused<<<dim3(EDIM, BDIM / 128), blk, 0, stream>>>(x1b, We1t, be1,
                                                           We2t, be2, vs);
  // out = sum_e ws[b,e] * vs[b,d,e]
  gate_reduce<<<dim3((BDIM * DDIM) / 256), blk, 0, stream>>>(wsg, vs, out);
}
